// PointModel_74354473828492
// MI455X (gfx1250) — matrix operand order repaired and checked by reading, not compiled, not run on hardware
//
#include <hip/hip_runtime.h>
#include <hip/hip_bf16.h>
#include <math.h>

typedef __attribute__((ext_vector_type(16))) _Float16 v16h;
typedef __attribute__((ext_vector_type(8)))  _Float16 hv8;
typedef __attribute__((ext_vector_type(8)))  float    v8f;

#define N_PTS   65536
#define M_BONES 1024
#define KWIN    64

// ---------------------------------------------------------------- prep
__global__ void prep_kernel(const float* __restrict__ x, float* __restrict__ sqx,
                            float* __restrict__ bones, float* __restrict__ sqb) {
  int i = blockIdx.x * blockDim.x + threadIdx.x;
  if (i < N_PTS) {
    float a = x[3*i], b = x[3*i+1], c = x[3*i+2];
    sqx[i] = a*a + b*b + c*c;
  }
  if (i < M_BONES) {
    int s = i * KWIN;
    float a = x[3*s], b = x[3*s+1], c = x[3*s+2];
    bones[3*i] = a; bones[3*i+1] = b; bones[3*i+2] = c;
    sqb[i] = a*a + b*b + c*c;
  }
}

__device__ __forceinline__ unsigned long long dist_key(float d, int idx) {
  unsigned u = __float_as_uint(d);
  u = (u & 0x80000000u) ? ~u : (u | 0x80000000u);
  return ((unsigned long long)u << 32) | (unsigned)idx;
}

// ------------------------------------------------- top-128 per bone (bitonic k-select)
__global__ void topk128_kernel(const float* __restrict__ x, const float* __restrict__ sqx,
                               const float* __restrict__ bones, const float* __restrict__ sqb,
                               int* __restrict__ widx) {
  int b = blockIdx.x;
  int t = threadIdx.x;                    // 128 threads
  __shared__ unsigned long long best[128];
  __shared__ unsigned long long cand[128];
  __shared__ float bx, by, bz, bs;
  if (t == 0) { bx = bones[3*b]; by = bones[3*b+1]; bz = bones[3*b+2]; bs = sqb[b]; }
  best[t] = ~0ull;
  __syncthreads();
  for (int base = 0; base < N_PTS; base += 128) {
    int j = base + t;
    float d = bs + sqx[j] - 2.f*(bx*x[3*j] + by*x[3*j+1] + bz*x[3*j+2]);
    cand[t] = dist_key(d, j);
    __syncthreads();
    for (int k2 = 2; k2 <= 128; k2 <<= 1) {          // bitonic sort cand ascending
      for (int jj = k2 >> 1; jj > 0; jj >>= 1) {
        int ixj = t ^ jj;
        if (ixj > t) {
          bool asc = ((t & k2) == 0);
          unsigned long long a0 = cand[t], a1 = cand[ixj];
          if ((a0 > a1) == asc) { cand[t] = a1; cand[ixj] = a0; }
        }
        __syncthreads();
      }
    }
    unsigned long long m0 = best[t], m1 = cand[127 - t];
    unsigned long long mg = (m0 < m1) ? m0 : m1;     // best(asc) vs cand reversed
    __syncthreads();
    best[t] = mg;
    __syncthreads();
    for (int jj = 64; jj > 0; jj >>= 1) {            // bitonic merge down
      int ixj = t ^ jj;
      if (ixj > t) {
        unsigned long long a0 = best[t], a1 = best[ixj];
        if (a0 > a1) { best[t] = a1; best[ixj] = a0; }
      }
      __syncthreads();
    }
  }
  widx[b*128 + t] = (int)(best[t] & 0xFFFFFFFFull);
}

// ---------------------------------------------------------------- aligned windows
__global__ void aligned_kernel(const float* __restrict__ x, const float* __restrict__ bones,
                               const int* __restrict__ widx,
                               float* __restrict__ a_in, float* __restrict__ a_out) {
  int idx = blockIdx.x * blockDim.x + threadIdx.x;
  if (idx >= M_BONES * KWIN) return;
  int b = idx >> 6, p = idx & 63;
  int ji = widx[b*128 + p];
  int jo = widx[b*128 + 2*p];
  #pragma unroll
  for (int c = 0; c < 3; ++c) {
    float bc = bones[3*b + c];
    a_in [idx*3 + c] = x[3*ji + c] - bc;
    a_out[idx*3 + c] = x[3*jo + c] - bc;
  }
}

// ------------------------------------------------- per-window kNN (full sort of 64)
__global__ void window_knn_kernel(const float* __restrict__ a, int* __restrict__ nbr) {
  int b = blockIdx.x;
  int p = threadIdx.x;                    // 64 threads
  __shared__ float px[64], py[64], pz[64], sq[64];
  {
    float X = a[(b*64+p)*3], Y = a[(b*64+p)*3+1], Z = a[(b*64+p)*3+2];
    px[p] = X; py[p] = Y; pz[p] = Z; sq[p] = X*X + Y*Y + Z*Z;
  }
  __syncthreads();
  unsigned long long key[64];
  float X = px[p], Y = py[p], Z = pz[p], S = sq[p];
  for (int q = 0; q < 64; ++q) {
    float d = S + sq[q] - 2.f*(X*px[q] + Y*py[q] + Z*pz[q]);
    key[q] = dist_key(d, q);
  }
  for (int i = 1; i < 64; ++i) {          // insertion sort ascending
    unsigned long long v = key[i];
    int j = i - 1;
    while (j >= 0 && key[j] > v) { key[j+1] = key[j]; --j; }
    key[j+1] = v;
  }
  int base = (b*64 + p) * 16;
  for (int k = 0; k < 16; ++k) {
    nbr[0*(N_PTS*16) + base + k] = (int)(key[k]   & 63ull);  // dil 1: top16
    nbr[1*(N_PTS*16) + base + k] = (int)(key[2*k] & 63ull);  // dil 2: top32 ::2
    nbr[2*(N_PTS*16) + base + k] = (int)(key[4*k] & 63ull);  // dil 4: top64 ::4
  }
}

// ---------------------------------------------------------------- bone-graph kNN
__global__ void bones_knn_kernel(const float* __restrict__ bones, const float* __restrict__ sqb,
                                 int* __restrict__ bnbr) {
  int p = blockIdx.x * blockDim.x + threadIdx.x;
  if (p >= M_BONES) return;
  unsigned long long best[32];
  for (int i = 0; i < 32; ++i) best[i] = ~0ull;
  float X = bones[3*p], Y = bones[3*p+1], Z = bones[3*p+2], S = sqb[p];
  for (int q = 0; q < M_BONES; ++q) {
    float d = S + sqb[q] - 2.f*(X*bones[3*q] + Y*bones[3*q+1] + Z*bones[3*q+2]);
    unsigned long long kk = dist_key(d, q);
    if (kk < best[31]) {
      int j = 31;
      while (j > 0 && best[j-1] > kk) { best[j] = best[j-1]; --j; }
      best[j] = kk;
    }
  }
  for (int k = 0; k < 8; ++k) {
    bnbr[(0*M_BONES + p)*8 + k] = (int)(best[k]   & 0xFFFFFFFFull);
    bnbr[(1*M_BONES + p)*8 + k] = (int)(best[2*k] & 0xFFFFFFFFull);
    bnbr[(2*M_BONES + p)*8 + k] = (int)(best[4*k] & 0xFFFFFFFFull);
  }
}

// ---------------------------------------------------------------- input projection
__global__ void inproj_kernel(const float* __restrict__ X, const float* __restrict__ Win,
                              const float* __restrict__ bin, float* __restrict__ h,
                              int R, int Kin) {
  int idx = blockIdx.x * blockDim.x + threadIdx.x;
  if (idx >= R * 64) return;
  int r = idx >> 6, c = idx & 63;
  float acc = bin[c];
  for (int k = 0; k < Kin; ++k) acc += X[r*Kin + k] * Win[k*64 + c];
  h[idx] = fmaxf(acc, 0.f);
}

// Wcat[c][o] = o<64 ? Wl[c][o] : Wl[64+c][o-64]   (so C = h@Wcat gives [h@Wl_top | h@Wl_bot])
__global__ void wcat_kernel(const float* __restrict__ Wl, float* __restrict__ Wcat) {
  int i = blockIdx.x * blockDim.x + threadIdx.x;
  if (i >= 64 * 128) return;
  int k = i >> 7, n = i & 127;
  Wcat[i] = (n < 64) ? Wl[k*64 + n] : Wl[(64 + k)*64 + (n - 64)];
}

// ------------------------- templated WMMA GEMM: C[R x N] = H[R x K] @ W[K x N] (+bias)(+relu)
// gridDim.x = R/64, blockDim = 256 (8 waves). Fully unrolled: KP/32 ksteps, (N/32) tiles/wave.
// LDS: H tile row-major [64][KP]; W tile TRANSPOSED [N][KP] so each B fragment is two
// contiguous 16B runs (2x ds_load_b128), in the ISA table k order 8*half+s, 16+8*half+(s-8).
template <int K, int N, int RELU>
__global__ __launch_bounds__(256) void wmma_gemm_kernel(const float* __restrict__ H,
                                                        const float* __restrict__ W,
                                                        const float* __restrict__ bias,
                                                        float* __restrict__ C) {
  constexpr int KP    = ((K + 31) / 32) * 32;
  constexpr int KSTEP = KP / 32;
  constexpr int NT    = N / 16;
  constexpr int ITERS = (4 * NT) / 8;      // tiles per wave (NT in {2,4,8})
  __shared__ __align__(16) _Float16 sH[64 * KP];
  __shared__ __align__(16) _Float16 sWt[N * KP];
  int tid  = threadIdx.x;
  int lane = tid & 31;
  int wave = tid >> 5;
  size_t rowbase = (size_t)blockIdx.x * 64;
  __builtin_prefetch(&H[rowbase * (size_t)K], 0, 1);
  #pragma unroll
  for (int i = tid; i < 64 * KP; i += 256) {
    int m = i / KP, k = i % KP;
    sH[i] = (k < K) ? (_Float16)H[(rowbase + m) * (size_t)K + k] : (_Float16)0.f;
  }
  #pragma unroll
  for (int i = tid; i < N * KP; i += 256) {
    int n = i / KP, k = i % KP;
    sWt[i] = (k < K) ? (_Float16)W[k * N + n] : (_Float16)0.f;
  }
  __syncthreads();
  int half = lane >> 4;
  int lidx = lane & 15;
  #pragma unroll
  for (int it = 0; it < ITERS; ++it) {
    int t  = wave + it * 8;
    int mt = t & 3, nt = t >> 2;
    const _Float16* hrow = sH  + (mt * 16 + lidx) * KP;
    const _Float16* wrow = sWt + (nt * 16 + lidx) * KP;
    v8f acc = {0.f, 0.f, 0.f, 0.f, 0.f, 0.f, 0.f, 0.f};
    #pragma unroll
    for (int ks = 0; ks < KSTEP; ++ks) {
      int kk = ks * 32;
      // A 16x32 f16 fragment: two contiguous 8-half runs at kk+8*half and kk+16+8*half
      hv8 alo = *(const hv8*)(hrow + kk + 8 * half);
      hv8 ahi = *(const hv8*)(hrow + kk + 16 + 8 * half);
      v16h af = __builtin_shufflevector(alo, ahi, 0,1,2,3,4,5,6,7,8,9,10,11,12,13,14,15);
      // B 32x16 f16 fragment: two contiguous 8-half runs at kk+8*half and kk+16+8*half (transposed store)
      hv8 blo = *(const hv8*)(wrow + kk + 8 * half);
      hv8 bhi = *(const hv8*)(wrow + kk + 16 + 8 * half);
      v16h bf = __builtin_shufflevector(blo, bhi, 0,1,2,3,4,5,6,7,8,9,10,11,12,13,14,15);
      acc = __builtin_amdgcn_wmma_f32_16x16x32_f16(false, af, false, bf,
                                                   (short)0, acc, false, false);
    }
    int col = nt * 16 + lidx;
    float bv = bias ? bias[col] : 0.f;
    #pragma unroll
    for (int r = 0; r < 8; ++r) {
      int row = mt * 16 + r + 8 * half;                 // D f32 layout: M = r + 8*half
      float v = acc[r] + bv;
      if (RELU) v = fmaxf(v, 0.f);
      C[(rowbase + row) * (size_t)N + col] = v;
    }
  }
}

// ---------------- edge combine: hnew = h + max_k relu(C[nbr,:64] - C[p,:64] + C[p,64:] + bl)
__global__ void edge_combine_kernel(const float* __restrict__ h, const float* __restrict__ C,
                                    const int* __restrict__ nbr, const float* __restrict__ bl,
                                    float* __restrict__ hout, int Pg, int nk, int R) {
  int idx = blockIdx.x * blockDim.x + threadIdx.x;
  if (idx >= R * 64) return;
  int r = idx >> 6, c = idx & 63;
  int gbase = (r / Pg) * Pg;
  float base = -C[(size_t)r*128 + c] + C[(size_t)r*128 + 64 + c] + bl[c];
  float mx = 0.f;                                   // relu => values >= 0
  for (int k = 0; k < nk; ++k) {
    int n = nbr[r*nk + k];
    float v = C[(size_t)(gbase + n)*128 + c] + base;
    mx = fmaxf(mx, v);
  }
  hout[idx] = h[idx] + mx;
}

// ---------------------------------------------------------------- pooling / bottleneck
__global__ void pool_kernel(const float* __restrict__ fin, const float* __restrict__ fout,
                            float* __restrict__ pooled) {
  int idx = blockIdx.x * blockDim.x + threadIdx.x;
  if (idx >= M_BONES * 128) return;
  int b = idx >> 7, c = idx & 127;
  const float* src = (c < 64) ? fin : fout;
  int cc = c & 63;
  float m = -3.402823466e38f;
  for (int p = 0; p < 64; ++p) m = fmaxf(m, src[(size_t)(b*64 + p)*64 + cc]);
  pooled[idx] = m;
}

__global__ void feat_kernel(const float* __restrict__ pooled, const float* __restrict__ W,
                            const float* __restrict__ b, float* __restrict__ q) {
  int idx = blockIdx.x * blockDim.x + threadIdx.x;
  if (idx >= M_BONES * 16) return;
  int m = idx >> 4, j = idx & 15;
  float acc = b[j];
  for (int c = 0; c < 128; ++c) acc += pooled[m*128 + c] * W[c*16 + j];
  q[idx] = acc;
}

// ------------------------------------ threefry2x32 uniform(-0.5,0.5) noise, key(42)=(0,42)
__device__ __forceinline__ unsigned rotl32(unsigned v, int s) { return (v << s) | (v >> (32 - s)); }

__global__ void noise_kernel(float* __restrict__ q) {
  int t = blockIdx.x * blockDim.x + threadIdx.x;
  if (t >= 8192) return;
  const unsigned k0 = 0u, k1 = 42u, k2 = 0u ^ 42u ^ 0x1BD11BDAu;
  unsigned x0 = (unsigned)t, x1 = (unsigned)(t + 8192);
  x0 += k0; x1 += k1;
  const int ra[4] = {13, 15, 26, 6}, rb[4] = {17, 29, 16, 24};
  #pragma unroll
  for (int i = 0; i < 4; ++i) { x0 += x1; x1 = rotl32(x1, ra[i]); x1 ^= x0; }
  x0 += k1; x1 += k2 + 1u;
  #pragma unroll
  for (int i = 0; i < 4; ++i) { x0 += x1; x1 = rotl32(x1, rb[i]); x1 ^= x0; }
  x0 += k2; x1 += k0 + 2u;
  #pragma unroll
  for (int i = 0; i < 4; ++i) { x0 += x1; x1 = rotl32(x1, ra[i]); x1 ^= x0; }
  x0 += k0; x1 += k1 + 3u;
  #pragma unroll
  for (int i = 0; i < 4; ++i) { x0 += x1; x1 = rotl32(x1, rb[i]); x1 ^= x0; }
  x0 += k1; x1 += k2 + 4u;
  #pragma unroll
  for (int i = 0; i < 4; ++i) { x0 += x1; x1 = rotl32(x1, ra[i]); x1 ^= x0; }
  x0 += k2; x1 += k0 + 5u;
  float u0 = __uint_as_float((x0 >> 9) | 0x3F800000u) - 1.f;
  float u1 = __uint_as_float((x1 >> 9) | 0x3F800000u) - 1.f;
  q[t]        += u0 - 0.5f;
  q[t + 8192] += u1 - 0.5f;
}

// ---------------------------------------------------------------- entropy bitrate
__global__ void bits_kernel(const float* __restrict__ q, const float* __restrict__ ent,
                            float* __restrict__ out_bit) {
  __shared__ float red[256];
  int tid = threadIdx.x;
  float sum = 0.f;
  for (int i = tid; i < M_BONES * 16; i += 256) {
    int m = i >> 4, j = i & 15;
    float mu = ent[m*32 + j];
    float s  = ent[m*32 + 16 + j];
    float sp = fmaxf(s, 0.f) + log1pf(expf(-fabsf(s)));   // softplus
    float sg = sp + 1e-6f;
    float inv = 1.f / (sg * 1.41421356237309515f);
    float v = q[i];
    float c1 = 0.5f * (1.f + erff((v + 0.5f - mu) * inv));
    float c0 = 0.5f * (1.f + erff((v - 0.5f - mu) * inv));
    float bb = -log2f((c1 - c0) + 1e-10f);
    sum += fminf(fmaxf(bb, 0.f), 50.f);
  }
  red[tid] = sum;
  __syncthreads();
  for (int s = 128; s > 0; s >>= 1) {
    if (tid < s) red[tid] += red[tid + s];
    __syncthreads();
  }
  if (tid == 0) out_bit[0] = red[0] / (float)N_PTS;
}

// ---------------------------------------------------------------- folding helpers
__global__ void fold_build_kernel(const float* __restrict__ f, const float* __restrict__ extra,
                                  float* __restrict__ out, int extraCols) {
  int cols = 64 + extraCols;
  int idx = blockIdx.x * blockDim.x + threadIdx.x;
  if (idx >= N_PTS * cols) return;
  int r = idx / cols, c = idx % cols;
  int m = r >> 6, k = r & 63;
  float v;
  if (c < 64) v = f[m*64 + c];
  else if (extraCols == 2) {
    int g = c - 64;
    int gi = (g == 0) ? (k & 7) : (k >> 3);               // meshgrid 'xy': (lin[k%8], lin[k/8])
    v = -1.f + 2.f * (float)gi / 7.f;
  } else v = extra[r*3 + (c - 64)];
  out[idx] = v;
}

__global__ void lin3_kernel(const float* __restrict__ h, const float* __restrict__ W,
                            const float* __restrict__ b, const float* __restrict__ bones_add,
                            float* __restrict__ out) {
  int idx = blockIdx.x * blockDim.x + threadIdx.x;
  if (idx >= N_PTS * 3) return;
  int r = idx / 3, n = idx % 3;
  float acc = b[n];
  for (int c = 0; c < 64; ++c) acc += h[(size_t)r*64 + c] * W[c*3 + n];
  if (bones_add) acc += bones_add[(r >> 6)*3 + n];
  out[idx] = acc;
}

// ================================================================ host driver
extern "C" void kernel_launch(void* const* d_in, const int* in_sizes, int n_in,
                              void* d_out, int out_size, void* d_ws, size_t ws_size,
                              hipStream_t stream) {
  (void)in_sizes; (void)n_in; (void)out_size; (void)ws_size;
  const float* x    = (const float*)d_in[0];
  const float* Psi  = (const float*)d_in[1];  // Win(192) bin(64) Wl(24576) bl(192) Wout(4096) bout(64)
  const float* Pso  = (const float*)d_in[2];
  const float* Plin = (const float*)d_in[3];  // W(2048) b(16)
  const float* Pent = (const float*)d_in[4];  // ... Wout(2048) bout(32)
  const float* Pstr = (const float*)d_in[5];  // Win(1024) bin(64) Wl(24576) bl(192) Wout(4096) bout(64)
  const float* Pf   = (const float*)d_in[6];  // W1(4224) b1(64) W2(192) b2(3) W3(4288) b3(64) W4(192) b4(3)
  float* out = (float*)d_out;

  float* base = (float*)d_ws;
  size_t o = 0;
  auto alloc = [&](size_t n) { float* p = base + o; o += n; return p; };
  float* sqx    = alloc(N_PTS);
  float* bones  = alloc(M_BONES*3);
  float* sqb    = alloc(M_BONES);
  int*   widx   = (int*)alloc(M_BONES*128);
  float* a_in   = alloc((size_t)N_PTS*3);
  float* a_out  = alloc((size_t)N_PTS*3);
  int*   nbr_in = (int*)alloc((size_t)3*N_PTS*16);
  int*   nbr_out= (int*)alloc((size_t)3*N_PTS*16);
  int*   bnbr   = (int*)alloc((size_t)3*M_BONES*8);
  float* hA     = alloc((size_t)N_PTS*64);
  float* hB     = alloc((size_t)N_PTS*64);
  float* Cbuf   = alloc((size_t)N_PTS*128);
  float* Wcat   = alloc(64*128);
  float* fin    = alloc((size_t)N_PTS*64);
  float* fout   = alloc((size_t)N_PTS*64);
  float* hbA    = alloc(M_BONES*64);
  float* hbB    = alloc(M_BONES*64);
  float* Cb     = alloc(M_BONES*128);
  float* pooled = alloc(M_BONES*128);
  float* qf     = alloc(M_BONES*16);
  float* entb   = alloc(M_BONES*32);
  float* fstr   = alloc(M_BONES*64);
  float* p1     = alloc((size_t)N_PTS*3);

  prep_kernel   <<<(N_PTS + 255)/256, 256, 0, stream>>>(x, sqx, bones, sqb);
  topk128_kernel<<<M_BONES, 128, 0, stream>>>(x, sqx, bones, sqb, widx);
  aligned_kernel<<<(N_PTS + 255)/256, 256, 0, stream>>>(x, bones, widx, a_in, a_out);
  window_knn_kernel<<<M_BONES, 64, 0, stream>>>(a_in,  nbr_in);
  window_knn_kernel<<<M_BONES, 64, 0, stream>>>(a_out, nbr_out);
  bones_knn_kernel <<<(M_BONES + 255)/256, 256, 0, stream>>>(bones, sqb, bnbr);

  auto resgcn = [&](const float* X, int Kin, const float* P,
                    size_t oWin, size_t oBin, size_t oWl, size_t oBl, size_t oWout, size_t oBout,
                    const int* nbr, int Pg, int nk, int R,
                    float* bufA, float* bufB, float* Cw, float* outp, int Nout) {
    int e = R * 64;
    inproj_kernel<<<(e + 255)/256, 256, 0, stream>>>(X, P + oWin, P + oBin, bufA, R, Kin);
    float* cur = bufA; float* nxt = bufB;
    for (int l = 0; l < 3; ++l) {
      wcat_kernel<<<32, 256, 0, stream>>>(P + oWl + (size_t)l*8192, Wcat);
      wmma_gemm_kernel<64,128,0><<<R/64, 256, 0, stream>>>(cur, Wcat, nullptr, Cw);
      edge_combine_kernel<<<(e + 255)/256, 256, 0, stream>>>(
          cur, Cw, nbr + (size_t)l*R*nk, P + oBl + (size_t)l*64, nxt, Pg, nk, R);
      float* tmp = cur; cur = nxt; nxt = tmp;
    }
    if (Nout == 32)
      wmma_gemm_kernel<64,32,0><<<R/64, 256, 0, stream>>>(cur, P + oWout, P + oBout, outp);
    else
      wmma_gemm_kernel<64,64,0><<<R/64, 256, 0, stream>>>(cur, P + oWout, P + oBout, outp);
  };

  // squeeze nets on 1024 windows x 64 pts (R=65536)
  resgcn(a_in,  3, Psi, 0,192,256,24832,25024,29120, nbr_in,  64, 16, N_PTS, hA, hB, Cbuf, fin,  64);
  resgcn(a_out, 3, Pso, 0,192,256,24832,25024,29120, nbr_out, 64, 16, N_PTS, hA, hB, Cbuf, fout, 64);

  pool_kernel<<<(M_BONES*128 + 255)/256, 256, 0, stream>>>(fin, fout, pooled);
  feat_kernel<<<(M_BONES*16 + 255)/256, 256, 0, stream>>>(pooled, Plin, Plin + 2048, qf);
  noise_kernel<<<32, 256, 0, stream>>>(qf);   // q = feat + U(-0.5,0.5)

  // entropy net on bones graph (R=1024)
  resgcn(bones, 3, Pent, 0,192,256,24832,25024,27072, bnbr, 1024, 8, M_BONES, hbA, hbB, Cb, entb, 32);
  bits_kernel<<<1, 256, 0, stream>>>(qf, entb, out + 196608);

  // structure net on q (R=1024)
  resgcn(qf, 16, Pstr, 0,1024,1088,25664,25856,29952, bnbr, 1024, 8, M_BONES, hbA, hbB, Cb, fstr, 64);

  // folding decoder (R=65536)
  fold_build_kernel<<<((size_t)N_PTS*66 + 255)/256, 256, 0, stream>>>(fstr, nullptr, Cbuf, 2);
  wmma_gemm_kernel<66,64,1><<<N_PTS/64, 256, 0, stream>>>(Cbuf, Pf + 0, Pf + 4224, hA);
  lin3_kernel<<<((size_t)N_PTS*3 + 255)/256, 256, 0, stream>>>(hA, Pf + 4288, Pf + 4480, nullptr, p1);
  fold_build_kernel<<<((size_t)N_PTS*67 + 255)/256, 256, 0, stream>>>(fstr, p1, Cbuf, 3);
  wmma_gemm_kernel<67,64,1><<<N_PTS/64, 256, 0, stream>>>(Cbuf, Pf + 4483, Pf + 8771, hA);
  lin3_kernel<<<((size_t)N_PTS*3 + 255)/256, 256, 0, stream>>>(hA, Pf + 8835, Pf + 9027, bones, out);
}
